// PolarityAwareConv_84877143703601
// MI455X (gfx1250) — compile-verified, matrix operand order checked by reading, not run on hardware
//
#include <hip/hip_runtime.h>
#include <hip/hip_bf16.h>

#define DN 256
#define F_EDGE 64
#define EDGE_ROW 65        // polarity + 64 features
#define NE 800000
#define NN 50000
#define LN_EPS 1e-5f

#if defined(__gfx1250__) && __has_builtin(__builtin_amdgcn_global_load_async_to_lds_b128)
#define HAVE_ASYNC 1
#else
#define HAVE_ASYNC 0
#endif

// ---------- types for WMMA ----------
typedef __attribute__((ext_vector_type(16))) __bf16 v16bf;
typedef __attribute__((ext_vector_type(8)))  float  v8f;
typedef __attribute__((ext_vector_type(4)))  int    v4i;

#if HAVE_ASYNC
typedef __attribute__((address_space(1))) v4i* gv4i_p;   // global (LangAS __device__)
typedef __attribute__((address_space(3))) v4i* lv4i_p;   // LDS (LangAS __shared__)
#endif

struct U4 { unsigned int a, b, c, d; };   // POD 128-bit chunk (union-safe)

union BF16x16 {
    v16bf v;
    unsigned int u[8];
    U4 q[2];
};

// native f32 -> bf16 (fptrunc; backend emits hw cvt on gfx1250)
static __device__ inline unsigned short f32_to_bf16(float f) {
    union { __bf16 b; unsigned short s; } u;
    u.b = (__bf16)f;
    return u.s;
}

// 16 contiguous f32 -> bf16x16 (scalar loads: unaligned-safe)
static __device__ inline BF16x16 cvt16_f32_s(const float* __restrict__ p) {
    BF16x16 r;
#pragma unroll
    for (int i = 0; i < 16; ++i) r.v[i] = (__bf16)p[i];
    return r;
}

// 16 contiguous f32 -> bf16x16 (16B-aligned fast path, works for LDS or global)
static __device__ inline BF16x16 cvt16_f32_v(const float* p) {
    BF16x16 r;
#pragma unroll
    for (int i = 0; i < 4; ++i) {
        float4 f = ((const float4*)p)[i];
        r.v[4 * i + 0] = (__bf16)f.x;
        r.v[4 * i + 1] = (__bf16)f.y;
        r.v[4 * i + 2] = (__bf16)f.z;
        r.v[4 * i + 3] = (__bf16)f.w;
    }
    return r;
}

// 16 contiguous bf16 (32B) from global or LDS
static __device__ inline BF16x16 load_bf16_16(const unsigned short* p) {
    BF16x16 r;
    r.q[0] = ((const U4*)p)[0];
    r.q[1] = ((const U4*)p)[1];
    return r;
}

static __device__ inline v8f wmma_bf16(v16bf a, v16bf b, v8f c) {
    // D = A(16x32 bf16) x B(32x16 bf16) + C(16x16 f32)
    return __builtin_amdgcn_wmma_f32_16x16x32_bf16(
        false, a, false, b, (short)0, c, false, false);
}

#define DS_FENCE() asm volatile("s_wait_dscnt 0" ::: "memory")

// ---------- kernel 1: aggr = x ----------
__global__ __launch_bounds__(256) void init_aggr_kernel(const float* __restrict__ x,
                                                        float* __restrict__ aggr) {
    int i = blockIdx.x * blockDim.x + threadIdx.x;   // float4 index
    if (i < (NN * DN) / 4) {
        ((float4*)aggr)[i] = ((const float4*)x)[i];
    }
}

// ---------- kernel 2: weights f32 -> bf16 ----------
// layout in wb: We1[16384] We2[65536] Wle[65536] Wg1[65536] Wg2[65536]
__global__ __launch_bounds__(256) void cvt_weights_kernel(
    const float* __restrict__ W_e1, const float* __restrict__ W_e2,
    const float* __restrict__ W_le, const float* __restrict__ W_g1,
    const float* __restrict__ W_g2, unsigned short* __restrict__ wb) {
    int i = blockIdx.x * blockDim.x + threadIdx.x;
    if (i >= 16384 + 4 * 65536) return;
    int j = i;
    float s;
    if (j < 16384) s = W_e1[j];
    else {
        j -= 16384;
        if (j < 65536) s = W_e2[j];
        else {
            j -= 65536;
            if (j < 65536) s = W_le[j];
            else {
                j -= 65536;
                if (j < 65536) s = W_g1[j];
                else           s = W_g2[j - 65536];
            }
        }
    }
    wb[i] = f32_to_bf16(s);
}

// ---------- kernel 3: edge pipeline (32 edges per wave, 1 wave per block) ----------
struct EdgeWaveSmem {
    unsigned short h1[32 * DN];     // relu(raw@We1^T+b1), bf16      (16KB)
    unsigned short gated[32 * DN];  // gated e, bf16                 (16KB)
    float pol[32];
    int   src[32];
    int   dst[32];
};

__global__ __launch_bounds__(32) void edge_kernel(
    const float* __restrict__ x,
    const long long* __restrict__ edge_index,
    const float* __restrict__ edge_attr,
    const unsigned short* __restrict__ Wb_e1, const float* __restrict__ b_e1,
    const unsigned short* __restrict__ Wb_e2, const float* __restrict__ b_e2,
    const unsigned short* __restrict__ Wb_le, const float* __restrict__ b_le,
    float* __restrict__ aggr) {
    __shared__ EdgeWaveSmem S;
    const int lane = threadIdx.x;       // 0..31, single wave
    const int eBase = blockIdx.x * 32;
    const int m  = lane & 15;   // row within 16-row tile
    const int kh = lane >> 4;   // K-half (A/B) or M-half (C/D) selector
    const int c  = m;           // column for B/C roles

    // per-edge metadata: each lane owns one of the 32 edges
    {
        int e = eBase + lane;
        float p = edge_attr[(size_t)e * EDGE_ROW];
        S.pol[lane] = fminf(fmaxf(p, 0.f), 1.f) + 0.01f;
        S.src[lane] = (int)edge_index[e];
        S.dst[lane] = (int)edge_index[(size_t)NE + e];
    }
    DS_FENCE();

    // speculative prefetch of next tile's rows
    if (eBase + 32 + lane < NE)
        __builtin_prefetch(&edge_attr[(size_t)(eBase + 32 + lane) * EDGE_ROW], 0, 1);

    // A operands for GEMM1: 2 M-tiles x 2 K-chunks of raw features (f32->bf16)
    BF16x16 A1[2][2];
#pragma unroll
    for (int mt = 0; mt < 2; ++mt) {
        const float* rr = &edge_attr[(size_t)(eBase + mt * 16 + m) * EDGE_ROW + 1 + kh * 16];
        A1[mt][0] = cvt16_f32_s(rr);
        A1[mt][1] = cvt16_f32_s(rr + 32);
    }

    // GEMM1: h1 = relu(raw @ We1^T + b_e1), K=64 ; B reused across both M-tiles
    for (int nt = 0; nt < 16; ++nt) {
        BF16x16 b0 = load_bf16_16(&Wb_e1[((size_t)(nt * 16 + c)) * F_EDGE + kh * 16]);
        BF16x16 b1 = load_bf16_16(&Wb_e1[((size_t)(nt * 16 + c)) * F_EDGE + 32 + kh * 16]);
        float bias = b_e1[nt * 16 + c];
#pragma unroll
        for (int mt = 0; mt < 2; ++mt) {
            v8f acc = {0.f, 0.f, 0.f, 0.f, 0.f, 0.f, 0.f, 0.f};
            acc = wmma_bf16(A1[mt][0].v, b0.v, acc);
            acc = wmma_bf16(A1[mt][1].v, b1.v, acc);
#pragma unroll
            for (int v = 0; v < 8; ++v) {
                int r = mt * 16 + v + 8 * kh;
                S.h1[r * DN + nt * 16 + c] = f32_to_bf16(fmaxf(acc[v] + bias, 0.f));
            }
        }
    }
    DS_FENCE();

    // GEMM2: e = h1 @ We2^T + b_e2 ; gated = e * (pol+0.01)
    // all 8 B K-chunks held in registers, reused across both M-tiles
    for (int nt = 0; nt < 16; ++nt) {
        BF16x16 B[8];
#pragma unroll
        for (int kc = 0; kc < 8; ++kc)
            B[kc] = load_bf16_16(&Wb_e2[((size_t)(nt * 16 + c)) * DN + kc * 32 + kh * 16]);
        float bias = b_e2[nt * 16 + c];
#pragma unroll
        for (int mt = 0; mt < 2; ++mt) {
            v8f acc = {0.f, 0.f, 0.f, 0.f, 0.f, 0.f, 0.f, 0.f};
#pragma unroll
            for (int kc = 0; kc < 8; ++kc) {
                BF16x16 a = load_bf16_16(&S.h1[(mt * 16 + m) * DN + kc * 32 + kh * 16]);
                acc = wmma_bf16(a.v, B[kc].v, acc);
            }
#pragma unroll
            for (int v = 0; v < 8; ++v) {
                int r = mt * 16 + v + 8 * kh;
                float g = (acc[v] + bias) * S.pol[r];
                S.gated[r * DN + nt * 16 + c] = f32_to_bf16(g);
            }
        }
    }
    DS_FENCE();

    // GEMM3: emb = gated @ Wle^T + b_le ; msg = relu(x[src]+emb) ; scatter-add
    for (int nt = 0; nt < 16; ++nt) {
        BF16x16 B[8];
#pragma unroll
        for (int kc = 0; kc < 8; ++kc)
            B[kc] = load_bf16_16(&Wb_le[((size_t)(nt * 16 + c)) * DN + kc * 32 + kh * 16]);
        float bias = b_le[nt * 16 + c];
#pragma unroll
        for (int mt = 0; mt < 2; ++mt) {
            v8f acc = {0.f, 0.f, 0.f, 0.f, 0.f, 0.f, 0.f, 0.f};
#pragma unroll
            for (int kc = 0; kc < 8; ++kc) {
                BF16x16 a = load_bf16_16(&S.gated[(mt * 16 + m) * DN + kc * 32 + kh * 16]);
                acc = wmma_bf16(a.v, B[kc].v, acc);
            }
#pragma unroll
            for (int v = 0; v < 8; ++v) {
                int r = mt * 16 + v + 8 * kh;
                int col = nt * 16 + c;
                float val = acc[v] + bias + x[(size_t)S.src[r] * DN + col];
                val = fmaxf(val, 0.f);
                unsafeAtomicAdd(&aggr[(size_t)S.dst[r] * DN + col], val);
            }
        }
    }
}

// ---------- kernel 4: node GEMM + LayerNorm + GEMM (16 nodes per wave) ----------
struct NodeWaveSmem {
    float xa[16 * DN];              // async-staged accumulator tile (16KB)
    float h[16 * DN];               // pre-LN activations            (16KB)
    unsigned short hn[16 * DN];     // relu(LN(h)) in bf16           (8KB)
};

__global__ __launch_bounds__(32) void node_kernel(
    const float* __restrict__ aggr,   // == x + segment_sum(msg)
    const unsigned short* __restrict__ Wb_g1, const float* __restrict__ b_g1,
    const float* __restrict__ ln_g, const float* __restrict__ ln_b,
    const unsigned short* __restrict__ Wb_g2, const float* __restrict__ b_g2,
    float* __restrict__ out) {
    __shared__ NodeWaveSmem S;
    const int lane = threadIdx.x;
    const int nodeBase = blockIdx.x * 16;   // 3125 blocks exactly
    const int m  = lane & 15;
    const int kh = lane >> 4;
    const int c  = m;

#if HAVE_ASYNC
    // bulk DMA the 16x256 f32 accumulator tile into LDS (ASYNCcnt path)
    {
        const char* gsrc = (const char*)&aggr[(size_t)nodeBase * DN];
        char* lbase = (char*)S.xa;
#pragma unroll
        for (int j = 0; j < 32; ++j) {
            int off = (j * 32 + lane) * 16;     // 32 iters x 32 lanes x 16B = 16KB
            __builtin_amdgcn_global_load_async_to_lds_b128(
                (gv4i_p)(gsrc + off),
                (lv4i_p)(lbase + off), 0, 0);
        }
    }
    asm volatile("s_wait_asynccnt 0" ::: "memory");
    const float* arows = S.xa;
#else
    const float* arows = &aggr[(size_t)nodeBase * DN];
#endif

    // A operand resident in registers: 16 rows x 256 K, bf16 (64 VGPRs)
    BF16x16 A[8];
#pragma unroll
    for (int kc = 0; kc < 8; ++kc)
        A[kc] = cvt16_f32_v(&arows[(size_t)m * DN + kc * 32 + kh * 16]);

    // GEMM1: h = (x+aggr) @ Wg1^T + b_g1
    for (int nt = 0; nt < 16; ++nt) {
        v8f acc = {0.f, 0.f, 0.f, 0.f, 0.f, 0.f, 0.f, 0.f};
#pragma unroll
        for (int kc = 0; kc < 8; ++kc) {
            BF16x16 b = load_bf16_16(&Wb_g1[((size_t)(nt * 16 + c)) * DN + kc * 32 + kh * 16]);
            acc = wmma_bf16(A[kc].v, b.v, acc);
        }
        float bias = b_g1[nt * 16 + c];
#pragma unroll
        for (int v = 0; v < 8; ++v)
            S.h[(v + 8 * kh) * DN + nt * 16 + c] = acc[v] + bias;
    }
    DS_FENCE();

    // LayerNorm: lane handles half of row m (half selected by kh)
    {
        const float* row = &S.h[m * DN + kh * 128];
        float sum = 0.f, sq = 0.f;
        for (int i = 0; i < 128; ++i) { float t = row[i]; sum += t; sq += t * t; }
        sum += __shfl_xor(sum, 16, 32);
        sq  += __shfl_xor(sq, 16, 32);
        float mean = sum * (1.f / 256.f);
        float var  = sq * (1.f / 256.f) - mean * mean;
        float rstd = rsqrtf(var + LN_EPS);
        for (int i = 0; i < 128; ++i) {
            int col = kh * 128 + i;
            float t = (S.h[m * DN + col] - mean) * rstd * ln_g[col] + ln_b[col];
            S.hn[m * DN + col] = f32_to_bf16(fmaxf(t, 0.f));
        }
    }
    DS_FENCE();

    // GEMM2: out = relu(hn) @ Wg2^T + b_g2
    for (int nt = 0; nt < 16; ++nt) {
        v8f acc = {0.f, 0.f, 0.f, 0.f, 0.f, 0.f, 0.f, 0.f};
#pragma unroll
        for (int kc = 0; kc < 8; ++kc) {
            BF16x16 a = load_bf16_16(&S.hn[m * DN + kc * 32 + kh * 16]);
            BF16x16 b = load_bf16_16(&Wb_g2[((size_t)(nt * 16 + c)) * DN + kc * 32 + kh * 16]);
            acc = wmma_bf16(a.v, b.v, acc);
        }
        float bias = b_g2[nt * 16 + c];
#pragma unroll
        for (int v = 0; v < 8; ++v)
            out[(size_t)(nodeBase + v + 8 * kh) * DN + nt * 16 + c] = acc[v] + bias;
    }
}

extern "C" void kernel_launch(void* const* d_in, const int* in_sizes, int n_in,
                              void* d_out, int out_size, void* d_ws, size_t ws_size,
                              hipStream_t stream) {
    const float*     x          = (const float*)d_in[0];
    const long long* edge_index = (const long long*)d_in[1];
    const float*     edge_attr  = (const float*)d_in[2];
    const float* W_e1 = (const float*)d_in[3];  const float* b_e1 = (const float*)d_in[4];
    const float* W_e2 = (const float*)d_in[5];  const float* b_e2 = (const float*)d_in[6];
    const float* W_le = (const float*)d_in[7];  const float* b_le = (const float*)d_in[8];
    const float* W_g1 = (const float*)d_in[9];  const float* b_g1 = (const float*)d_in[10];
    const float* ln_g = (const float*)d_in[11]; const float* ln_b = (const float*)d_in[12];
    const float* W_g2 = (const float*)d_in[13]; const float* b_g2 = (const float*)d_in[14];
    float* out = (float*)d_out;

    // workspace layout: [ aggr f32 (NN*DN) | bf16 weights ]
    float* aggr = (float*)d_ws;
    unsigned short* wb = (unsigned short*)((char*)d_ws + (size_t)NN * DN * sizeof(float));
    unsigned short* Wb_e1 = wb;
    unsigned short* Wb_e2 = wb + 16384;
    unsigned short* Wb_le = wb + 16384 + 65536;
    unsigned short* Wb_g1 = wb + 16384 + 2 * 65536;
    unsigned short* Wb_g2 = wb + 16384 + 3 * 65536;

    init_aggr_kernel<<<(NN * DN / 4 + 255) / 256, 256, 0, stream>>>(x, aggr);
    cvt_weights_kernel<<<(16384 + 4 * 65536 + 255) / 256, 256, 0, stream>>>(
        W_e1, W_e2, W_le, W_g1, W_g2, wb);
    edge_kernel<<<NE / 32, 32, 0, stream>>>(
        x, edge_index, edge_attr, Wb_e1, b_e1, Wb_e2, b_e2, Wb_le, b_le, aggr);
    node_kernel<<<NN / 16, 32, 0, stream>>>(
        aggr, Wb_g1, b_g1, ln_g, ln_b, Wb_g2, b_g2, out);

    (void)in_sizes; (void)n_in; (void)out_size; (void)ws_size;
}
